// TransformerDecoderLayer_19971597926667
// MI455X (gfx1250) — matrix operand order repaired and checked by reading, not compiled, not run on hardware
//
#include <hip/hip_runtime.h>

#define BB   8
#define TT   64
#define DD   512
#define HH   8
#define DH   64
#define DFF  2048
#define NEe  128
#define NSs  32
#define PLl  64
#define MAXP 5
#define NEGV -1e18f

typedef _Float16 h8  __attribute__((ext_vector_type(8)));
typedef _Float16 h16 __attribute__((ext_vector_type(16)));
typedef float    f8  __attribute__((ext_vector_type(8)));

union HF { h16 v; h8 h[2]; };

// Load one 16x32 f16 fragment (A-layout, K-contig rows; B uses ld_fragB) reduces to two
// 16-byte contiguous chunks per lane at +0 and +stride_half).
__device__ __forceinline__ h16 ld_frag(const _Float16* p, int second_off) {
  HF f;
  f.h[0] = *(const h8*)(p);
  f.h[1] = *(const h8*)(p + second_off);
  return f.v;
}

__device__ __forceinline__ h16 ld_fragB(const _Float16* p, int ld) {
  h16 v;
  for (int s = 0; s < 16; ++s) v[s] = p[(size_t)(s < 8 ? s : s + 8) * ld];
  return v;
}
__device__ __forceinline__ void wmma_store(
    f8 acc, int row0, int col, size_t zc, int ldc,
    const float* __restrict__ bias, const float* __restrict__ resid,
    float rscale, float* __restrict__ Cf, _Float16* __restrict__ Ch, int relu)
{
  const float bv = bias ? bias[col] : 0.f;
  for (int j = 0; j < 8; ++j) {
    const size_t off = zc + (size_t)(row0 + j) * ldc + col;
    float v = acc[j] + bv;
    if (resid) v += rscale * resid[off];
    if (relu)  v  = fmaxf(v, 0.f);
    if (Cf) Cf[off] = v;
    if (Ch) Ch[off] = (_Float16)v;
  }
}

// ---------------------------------------------------------------------------
// Register-blocked WMMA GEMM: wave tile 32x32 (2x2 wmma), WG = 4 waves (2x2)
// -> 64x64 tile. C[z] = A[z](MxK, K-contig) x Bt[z](KxN, N-contig) +bias+resid
// Requires M%64==0, N%64==0, K%32==0.
// ---------------------------------------------------------------------------
__global__ __launch_bounds__(128) void k_gemm_wmma64(
    const _Float16* __restrict__ A, const _Float16* __restrict__ Bt,
    const float* __restrict__ bias, const float* __restrict__ resid,
    float rscale, float* __restrict__ Cf, _Float16* __restrict__ Ch,
    int K, int lda, int ldb, int ldc, int sA, int sB, int sC, int relu)
{
  const int tid  = threadIdx.x;
  const int lane = tid & 31;
  const int wid  = tid >> 5;
  const int m0 = blockIdx.y * 64 + (wid >> 1) * 32;
  const int n0 = blockIdx.x * 64 + (wid & 1) * 32;
  const int z  = blockIdx.z;
  const int am = lane & 15;
  const int g  = lane >> 4;
  const _Float16* ap0 = A  + (size_t)z * sA + (size_t)(m0 + am) * lda + g * 8;
  const _Float16* ap1 = ap0 + (size_t)16 * lda;
  const _Float16* bp0 = Bt + (size_t)z * sB + (size_t)(g * 8) * ldb + n0 + am;
  const _Float16* bp1 = bp0 + 16;
  f8 acc00 = {0.f,0.f,0.f,0.f,0.f,0.f,0.f,0.f};
  f8 acc01 = acc00, acc10 = acc00, acc11 = acc00;
  for (int k = 0; k < K; k += 32) {
    const h16 a0 = ld_frag(ap0, 16);
    const h16 a1 = ld_frag(ap1, 16);
    const h16 b0 = ld_fragB(bp0, ldb);
    const h16 b1 = ld_fragB(bp1, ldb);
    __builtin_prefetch(ap0 + 32, 0, 1);                // global_prefetch_b8
    __builtin_prefetch(ap1 + 32, 0, 1);
    __builtin_prefetch(bp0 + (size_t)32 * ldb, 0, 1);
    __builtin_prefetch(bp1 + (size_t)32 * ldb, 0, 1);
    acc00 = __builtin_amdgcn_wmma_f32_16x16x32_f16(false, a0, false, b0,
                                                   (short)0, acc00, false, false);
    acc01 = __builtin_amdgcn_wmma_f32_16x16x32_f16(false, a0, false, b1,
                                                   (short)0, acc01, false, false);
    acc10 = __builtin_amdgcn_wmma_f32_16x16x32_f16(false, a1, false, b0,
                                                   (short)0, acc10, false, false);
    acc11 = __builtin_amdgcn_wmma_f32_16x16x32_f16(false, a1, false, b1,
                                                   (short)0, acc11, false, false);
    ap0 += 32; ap1 += 32;
    bp0 += (size_t)32 * ldb; bp1 += (size_t)32 * ldb;
  }
  const size_t zc = (size_t)z * sC;
  const int rb = m0 + g * 8;
  const int c0 = n0 + am;
  wmma_store(acc00, rb,      c0,      zc, ldc, bias, resid, rscale, Cf, Ch, relu);
  wmma_store(acc01, rb,      c0 + 16, zc, ldc, bias, resid, rscale, Cf, Ch, relu);
  wmma_store(acc10, rb + 16, c0,      zc, ldc, bias, resid, rscale, Cf, Ch, relu);
  wmma_store(acc11, rb + 16, c0 + 16, zc, ldc, bias, resid, rscale, Cf, Ch, relu);
}

// --------------- fallback WMMA GEMM (32x32 WG tile, any M,N %32) -----------
__global__ __launch_bounds__(128) void k_gemm_wmma32(
    const _Float16* __restrict__ A, const _Float16* __restrict__ Bt,
    const float* __restrict__ bias, const float* __restrict__ resid,
    float rscale, float* __restrict__ Cf, _Float16* __restrict__ Ch,
    int K, int lda, int ldb, int ldc, int sA, int sB, int sC, int relu)
{
  const int tid  = threadIdx.x;
  const int lane = tid & 31;
  const int wid  = tid >> 5;
  const int m0 = blockIdx.y * 32 + (wid >> 1) * 16;
  const int n0 = blockIdx.x * 32 + (wid & 1) * 16;
  const int z  = blockIdx.z;
  const int am = lane & 15;
  const int g  = lane >> 4;
  const _Float16* ap = A  + (size_t)z * sA + (size_t)(m0 + am) * lda + g * 8;
  const _Float16* bp = Bt + (size_t)z * sB + (size_t)(g * 8) * ldb + n0 + am;
  f8 acc = {0.f,0.f,0.f,0.f,0.f,0.f,0.f,0.f};
  for (int k = 0; k < K; k += 32) {
    const h16 a = ld_frag(ap, 16);
    const h16 b = ld_fragB(bp, ldb);
    acc = __builtin_amdgcn_wmma_f32_16x16x32_f16(false, a, false, b,
                                                 (short)0, acc, false, false);
    ap += 32;
    bp += (size_t)32 * ldb;
  }
  wmma_store(acc, m0 + g * 8, n0 + am, (size_t)z * sC, ldc,
             bias, resid, rscale, Cf, Ch, relu);
}

// --------------------------- LayerNorm (width 512) -------------------------
__global__ __launch_bounds__(256) void k_layernorm(
    const float* __restrict__ in, const float* __restrict__ gam,
    const float* __restrict__ bet, float* __restrict__ outf,
    _Float16* __restrict__ outh, float prescale)
{
  __shared__ float r1[256], r2[256];
  const int row = blockIdx.x, tid = threadIdx.x;
  const float* x = in + (size_t)row * DD;
  const float v0 = prescale * x[tid];
  const float v1 = prescale * x[tid + 256];
  r1[tid] = v0 + v1;
  r2[tid] = v0 * v0 + v1 * v1;
  __syncthreads();
  for (int o = 128; o > 0; o >>= 1) {
    if (tid < o) { r1[tid] += r1[tid + o]; r2[tid] += r2[tid + o]; }
    __syncthreads();
  }
  const float mean = r1[0] * (1.0f / DD);
  const float var  = r2[0] * (1.0f / DD) - mean * mean;
  const float rstd = rsqrtf(var + 1e-6f);
  const float o0 = (v0 - mean) * rstd * gam[tid] + bet[tid];
  const float o1 = (v1 - mean) * rstd * gam[tid + 256] + bet[tid + 256];
  const size_t base = (size_t)row * DD;
  if (outf) { outf[base + tid] = o0; outf[base + tid + 256] = o1; }
  if (outh) { outh[base + tid] = (_Float16)o0; outh[base + tid + 256] = (_Float16)o1; }
}

// --------------------------- f32 -> f16 converts ---------------------------
__global__ void k_conv_h(const float* __restrict__ s, _Float16* __restrict__ d, int n)
{
  const int i = blockIdx.x * 256 + threadIdx.x;
  if (i < n) d[i] = (_Float16)s[i];
}

// f32 [batch,R,C] -> f16 [batch,C,R]   (weight / edge transpose for B operand)
__global__ void k_conv_hT(const float* __restrict__ s, _Float16* __restrict__ d,
                          int R, int C)
{
  const int bz = blockIdx.y;
  const int i  = blockIdx.x * 256 + threadIdx.x;
  const int n  = R * C;
  if (i >= n) return;
  const int r = i / C, c = i - r * C;
  d[(size_t)bz * n + (size_t)c * R + r] = (_Float16)s[(size_t)bz * n + i];
}

// ------------------ head pack / unpack for self-attention ------------------
// src f16 [B,T,H*DH] -> dst [B,H,T,DH] (tr=0) or [B,H,DH,T] (tr=1), scaled
__global__ void k_pack_heads(const _Float16* __restrict__ s, _Float16* __restrict__ d,
                             float scale, int tr)
{
  const int i = blockIdx.x * 256 + threadIdx.x;          // 262144 elements
  const int dd = i & 63, t = (i >> 6) & 63, h = (i >> 12) & 7, b = i >> 15;
  const float v = (float)s[(size_t)((b * TT + t) * HH + h) * DH + dd] * scale;
  const size_t o = tr ? (size_t)(((b * HH + h) * DH + dd) * TT + t)
                      : (size_t)(((b * HH + h) * TT + t) * DH + dd);
  d[o] = (_Float16)v;
}

// src [B,H,T,DH] -> dst [B,T,H*DH]
__global__ void k_unpack_heads(const _Float16* __restrict__ s, _Float16* __restrict__ d)
{
  const int i = blockIdx.x * 256 + threadIdx.x;
  const int dd = i & 63, h = (i >> 6) & 7, t = (i >> 9) & 63, b = i >> 15;
  d[i] = s[(size_t)(((b * HH + h) * TT + t) * DH + dd)];
}

// ----------------- softmax for self-attn (causal + pad mask) ---------------
__global__ __launch_bounds__(256) void k_softmax_self(
    const float* __restrict__ sc, const unsigned char* __restrict__ pad,
    _Float16* __restrict__ pr)
{
  const int tid = threadIdx.x, wid = tid >> 5, lane = tid & 31;
  const int row = blockIdx.x * 8 + wid;                 // [B,H,T] rows = 4096
  const int b  = row >> 9;
  const int qi = row & 63;
  const unsigned char* pm = pad + (size_t)(b * TT + qi) * TT;
  const size_t base = (size_t)row * TT;
  const int j0 = lane, j1 = lane + 32;
  float v0 = (j0 > qi || pm[j0]) ? NEGV : sc[base + j0];
  float v1 = (j1 > qi || pm[j1]) ? NEGV : sc[base + j1];
  float m = fmaxf(v0, v1);
  for (int o = 16; o > 0; o >>= 1) m = fmaxf(m, __shfl_xor(m, o, 32));
  const float e0 = __expf(v0 - m), e1 = __expf(v1 - m);
  float s = e0 + e1;
  for (int o = 16; o > 0; o >>= 1) s += __shfl_xor(s, o, 32);
  const float inv = 1.0f / s;
  pr[base + j0] = (_Float16)(e0 * inv);
  pr[base + j1] = (_Float16)(e1 * inv);
}

// --------------------- softmax over entities (width 128) -------------------
__global__ __launch_bounds__(256) void k_softmax_ent(
    const float* __restrict__ sc, const unsigned char* __restrict__ em,
    _Float16* __restrict__ pr)
{
  const int tid = threadIdx.x, wid = tid >> 5, lane = tid & 31;
  const int row = blockIdx.x * 8 + wid;                 // B*T rows = 512
  const int b = row >> 6;
  const size_t base = (size_t)row * NEe;
  float v[4], e[4];
  float m = -3.0e38f;
  for (int c = 0; c < 4; ++c) {
    const int j = lane + 32 * c;
    v[c] = em[b * NEe + j] ? NEGV : sc[base + j];
    m = fmaxf(m, v[c]);
  }
  for (int o = 16; o > 0; o >>= 1) m = fmaxf(m, __shfl_xor(m, o, 32));
  float s = 0.f;
  for (int c = 0; c < 4; ++c) { e[c] = __expf(v[c] - m); s += e[c]; }
  for (int o = 16; o > 0; o >>= 1) s += __shfl_xor(s, o, 32);
  const float inv = 1.0f / s;
  for (int c = 0; c < 4; ++c) pr[base + lane + 32 * c] = (_Float16)(e[c] * inv);
}

// ------------- stable top-5 paragraph selection (argsort desc) -------------
__global__ void k_top5(const float* __restrict__ ap, int* __restrict__ idx)
{
  const int bt = blockIdx.x * blockDim.x + threadIdx.x;
  if (bt >= BB * TT) return;
  const float* v = ap + (size_t)bt * NSs;
  unsigned chosen = 0u;
  for (int j = 0; j < MAXP; ++j) {
    float best = -3.4e38f; int bi = 0;
    for (int s = 0; s < NSs; ++s) {
      if (chosen & (1u << s)) continue;
      const float x = v[s];
      if (x > best) { best = x; bi = s; }   // strict > == stable argsort
    }
    chosen |= 1u << bi;
    idx[bt * MAXP + j] = bi;
  }
}

// --------- gathered cross-attention over 5x64 pre-projected K/V rows -------
__global__ __launch_bounds__(320) void k_cross_attn(
    const _Float16* __restrict__ q, const _Float16* __restrict__ kall,
    const _Float16* __restrict__ vall, const int* __restrict__ src,
    const int* __restrict__ idx, _Float16* __restrict__ ctx)
{
  __shared__ float qs[DD];
  __shared__ float sc[HH * 320];
  __shared__ int   rowb[320];
  __shared__ int   pidx[MAXP];
  const int bt = blockIdx.x;
  const int b  = bt >> 6;
  const int tid = threadIdx.x;
  for (int i = tid; i < DD; i += 320) qs[i] = (float)q[(size_t)bt * DD + i];
  if (tid < MAXP) pidx[tid] = idx[bt * MAXP + tid];
  __syncthreads();
  {
    const int j = tid >> 6, i = tid & 63;
    const int p = pidx[j];
    const int r = (b * NSs + p) * PLl + i;
    rowb[tid] = r * DD;
    const bool masked = (src[r] == 0);                  // PAD == 0
    const _Float16* kr = kall + (size_t)r * DD;
    for (int h = 0; h < HH; ++h) {
      float s = 0.f;
      for (int d = 0; d < DH; ++d) s += qs[h * DH + d] * (float)kr[h * DH + d];
      sc[h * 320 + tid] = masked ? NEGV : s * 0.125f;   // 1/sqrt(64)
    }
  }
  __syncthreads();
  const int wid = tid >> 5, lane = tid & 31;
  if (wid < HH) {                                       // one wave per head
    float m = -3.0e38f;
    for (int kk = lane; kk < 320; kk += 32) m = fmaxf(m, sc[wid * 320 + kk]);
    for (int o = 16; o > 0; o >>= 1) m = fmaxf(m, __shfl_xor(m, o, 32));
    float eb[10]; int c = 0; float s = 0.f;
    for (int kk = lane; kk < 320; kk += 32) {
      const float e = __expf(sc[wid * 320 + kk] - m);
      eb[c++] = e; s += e;
    }
    for (int o = 16; o > 0; o >>= 1) s += __shfl_xor(s, o, 32);
    const float inv = 1.0f / s;
    c = 0;
    for (int kk = lane; kk < 320; kk += 32) sc[wid * 320 + kk] = eb[c++] * inv;
  }
  __syncthreads();
  for (int o = tid; o < DD; o += 320) {
    const int h = o >> 6;
    float acc = 0.f;
    for (int kk = 0; kk < 320; ++kk)
      acc += sc[h * 320 + kk] * (float)vall[(size_t)rowb[kk] + o];
    ctx[(size_t)bt * DD + o] = (_Float16)acc;
  }
}

// ===========================================================================
extern "C" void kernel_launch(void* const* d_in, const int* in_sizes, int n_in,
                              void* d_out, int out_size, void* d_ws, size_t ws_size,
                              hipStream_t stream) {
  (void)in_sizes; (void)n_in; (void)out_size; (void)ws_size;
  const float* inputs       = (const float*)d_in[0];
  const int*   src          = (const int*)d_in[1];
  const float* src_features = (const float*)d_in[2];
  const float* ent_feature  = (const float*)d_in[3];
  const unsigned char* ent_mask = (const unsigned char*)d_in[4];
  const float* edge         = (const float*)d_in[5];
  const unsigned char* tgt_pad = (const unsigned char*)d_in[6];
  const float *sa_wq=(const float*)d_in[7],  *sa_bq=(const float*)d_in[8];
  const float *sa_wk=(const float*)d_in[9],  *sa_bk=(const float*)d_in[10];
  const float *sa_wv=(const float*)d_in[11], *sa_bv=(const float*)d_in[12];
  const float *sa_wo=(const float*)d_in[13], *sa_bo=(const float*)d_in[14];
  const float *ca_wq=(const float*)d_in[15], *ca_bq=(const float*)d_in[16];
  const float *ca_wk=(const float*)d_in[17], *ca_bk=(const float*)d_in[18];
  const float *ca_wv=(const float*)d_in[19], *ca_bv=(const float*)d_in[20];
  const float *ca_wo=(const float*)d_in[21], *ca_bo=(const float*)d_in[22];
  const float *e_wi=(const float*)d_in[23],  *e_bi=(const float*)d_in[24];
  const float *e_we=(const float*)d_in[25],  *e_be=(const float*)d_in[26];
  const float *f_w1=(const float*)d_in[27],  *f_b1=(const float*)d_in[28];
  const float *f_w2=(const float*)d_in[29],  *f_b2=(const float*)d_in[30];
  const float *ln1_g=(const float*)d_in[31], *ln1_b=(const float*)d_in[32];
  const float *ln2_g=(const float*)d_in[33], *ln2_b=(const float*)d_in[34];
  const float *fln_g=(const float*)d_in[35], *fln_b=(const float*)d_in[36];
  float* out = (float*)d_out;                 // [output | input_norm]

  // ---- workspace allocator ----
  char* w = (char*)d_ws; size_t off = 0;
  auto alloc = [&](size_t bytes) -> void* {
    off = (off + 255) & ~(size_t)255;
    void* p = w + off; off += bytes; return p;
  };
  const size_t BT = (size_t)BB * TT;          // 512
  const size_t E  = BT * DD;                  // 262144
  _Float16* wt_sa_q=(_Float16*)alloc(DD*DD*2); _Float16* wt_sa_k=(_Float16*)alloc(DD*DD*2);
  _Float16* wt_sa_v=(_Float16*)alloc(DD*DD*2); _Float16* wt_sa_o=(_Float16*)alloc(DD*DD*2);
  _Float16* wt_ca_q=(_Float16*)alloc(DD*DD*2); _Float16* wt_ca_k=(_Float16*)alloc(DD*DD*2);
  _Float16* wt_ca_v=(_Float16*)alloc(DD*DD*2); _Float16* wt_ca_o=(_Float16*)alloc(DD*DD*2);
  _Float16* wt_e_wi=(_Float16*)alloc(DD*DD*2); _Float16* wt_e_we=(_Float16*)alloc(DD*DD*2);
  _Float16* wt_f1  =(_Float16*)alloc((size_t)DD*DFF*2);
  _Float16* wt_f2  =(_Float16*)alloc((size_t)DFF*DD*2);
  _Float16* h_in_norm=(_Float16*)alloc(E*2);
  _Float16* h_tq=(_Float16*)alloc(E*2); _Float16* h_tk=(_Float16*)alloc(E*2);
  _Float16* h_tv=(_Float16*)alloc(E*2);
  _Float16* h_qp=(_Float16*)alloc(E*2); _Float16* h_kT=(_Float16*)alloc(E*2);
  _Float16* h_vp=(_Float16*)alloc(E*2);
  float*    sc_self=(float*)alloc(E*4);
  _Float16* h_probs=(_Float16*)alloc(E*2);
  _Float16* h_ctx_hd=(_Float16*)alloc(E*2);
  _Float16* h_ctx=(_Float16*)alloc(E*2);
  float*    queryB=(float*)alloc(E*4);
  _Float16* h_qnorm=(_Float16*)alloc(E*2);
  _Float16* h_entf=(_Float16*)alloc((size_t)BB*NEe*DD*2);
  _Float16* h_eq=(_Float16*)alloc(E*2);
  float*    ek_f32=(float*)alloc((size_t)BB*NEe*DD*4);
  _Float16* h_ekT=(_Float16*)alloc((size_t)BB*DD*NEe*2);
  float*    sc_ent=(float*)alloc((size_t)BT*NEe*4);
  _Float16* h_aent=(_Float16*)alloc((size_t)BT*NEe*2);
  _Float16* h_edgeT=(_Float16*)alloc((size_t)BB*NEe*NSs*2);
  float*    a_para=(float*)alloc((size_t)BT*NSs*4);
  int*      idx_top=(int*)alloc((size_t)BT*MAXP*4);
  const size_t SR = (size_t)BB*NSs*PLl;       // 16384 source rows
  _Float16* h_srcf=(_Float16*)alloc(SR*DD*2);
  _Float16* h_kall=(_Float16*)alloc(SR*DD*2);
  _Float16* h_vall=(_Float16*)alloc(SR*DD*2);
  _Float16* h_cq=(_Float16*)alloc(E*2);
  _Float16* h_cctx=(_Float16*)alloc(E*2);
  float*    mid=(float*)alloc(E*4);
  _Float16* h_lnx=(_Float16*)alloc(E*2);
  _Float16* h_h1=(_Float16*)alloc((size_t)BT*DFF*2);

  auto gemm = [&](const _Float16* A, const _Float16* Bt, const float* bias,
                  const float* resid, float rs, float* Cf, _Float16* Ch,
                  int M, int N, int K, int lda, int ldb, int ldc,
                  int sA, int sB, int sC, int batch, int relu) {
    if ((M % 64 == 0) && (N % 64 == 0)) {
      dim3 g(N / 64, M / 64, batch);
      k_gemm_wmma64<<<g, 128, 0, stream>>>(A, Bt, bias, resid, rs, Cf, Ch,
                                           K, lda, ldb, ldc, sA, sB, sC, relu);
    } else {
      dim3 g(N / 32, M / 32, batch);
      k_gemm_wmma32<<<g, 128, 0, stream>>>(A, Bt, bias, resid, rs, Cf, Ch,
                                           K, lda, ldb, ldc, sA, sB, sC, relu);
    }
  };
  auto convT = [&](const float* s, _Float16* d, int batch, int R, int C) {
    dim3 g((R * C + 255) / 256, batch);
    k_conv_hT<<<g, 256, 0, stream>>>(s, d, R, C);
  };
  auto conv = [&](const float* s, _Float16* d, int n) {
    k_conv_h<<<(n + 255) / 256, 256, 0, stream>>>(s, d, n);
  };

  // ---- f16 conversions (weights pre-transposed for the WMMA B operand) ----
  convT(sa_wq, wt_sa_q, 1, DD, DD);  convT(sa_wk, wt_sa_k, 1, DD, DD);
  convT(sa_wv, wt_sa_v, 1, DD, DD);  convT(sa_wo, wt_sa_o, 1, DD, DD);
  convT(ca_wq, wt_ca_q, 1, DD, DD);  convT(ca_wk, wt_ca_k, 1, DD, DD);
  convT(ca_wv, wt_ca_v, 1, DD, DD);  convT(ca_wo, wt_ca_o, 1, DD, DD);
  convT(e_wi,  wt_e_wi, 1, DD, DD);  convT(e_we,  wt_e_we, 1, DD, DD);
  convT(f_w1,  wt_f1,   1, DFF, DD); convT(f_w2,  wt_f2,   1, DD, DFF);
  convT(edge,  h_edgeT, BB, NSs, NEe);                 // [b,s,e] -> [b,e,s]
  conv(src_features, h_srcf, (int)(SR * DD));
  conv(ent_feature,  h_entf, BB * NEe * DD);

  // ---- LN1 -> input_norm (second output) + f16 copy ----
  k_layernorm<<<(int)BT, 256, 0, stream>>>(inputs, ln1_g, ln1_b,
                                           out + E, h_in_norm, 1.0f);

  // ---- self-attention ----
  gemm(h_in_norm, wt_sa_q, sa_bq, nullptr, 0.f, nullptr, h_tq,
       512, 512, 512, 512, 512, 512, 0, 0, 0, 1, 0);
  gemm(h_in_norm, wt_sa_k, sa_bk, nullptr, 0.f, nullptr, h_tk,
       512, 512, 512, 512, 512, 512, 0, 0, 0, 1, 0);
  gemm(h_in_norm, wt_sa_v, sa_bv, nullptr, 0.f, nullptr, h_tv,
       512, 512, 512, 512, 512, 512, 0, 0, 0, 1, 0);
  k_pack_heads<<<1024, 256, 0, stream>>>(h_tq, h_qp, 0.125f, 0); // q/sqrt(DH)
  k_pack_heads<<<1024, 256, 0, stream>>>(h_tk, h_kT, 1.0f, 1);   // [B,H,DH,T]
  k_pack_heads<<<1024, 256, 0, stream>>>(h_tv, h_vp, 1.0f, 0);   // [B,H,T,DH]
  gemm(h_qp, h_kT, nullptr, nullptr, 0.f, sc_self, nullptr,      // scores
       64, 64, 64, 64, 64, 64, 4096, 4096, 4096, BB * HH, 0);
  k_softmax_self<<<512, 256, 0, stream>>>(sc_self, tgt_pad, h_probs);
  gemm(h_probs, h_vp, nullptr, nullptr, 0.f, nullptr, h_ctx_hd,  // attn @ V
       64, 64, 64, 64, 64, 64, 4096, 4096, 4096, BB * HH, 0);
  k_unpack_heads<<<1024, 256, 0, stream>>>(h_ctx_hd, h_ctx);
  gemm(h_ctx, wt_sa_o, sa_bo, inputs, 1.0f, queryB, nullptr,     // + residual
       512, 512, 512, 512, 512, 512, 0, 0, 0, 1, 0);
  k_layernorm<<<(int)BT, 256, 0, stream>>>(queryB, ln2_g, ln2_b,
                                           nullptr, h_qnorm, 1.0f);

  // ---- hierarchical entity attention ----
  gemm(h_qnorm, wt_e_wi, e_bi, nullptr, 0.f, nullptr, h_eq,
       512, 512, 512, 512, 512, 512, 0, 0, 0, 1, 0);
  gemm(h_entf, wt_e_we, e_be, nullptr, 0.f, ek_f32, nullptr,
       BB * NEe, 512, 512, 512, 512, 512, 0, 0, 0, 1, 0);
  convT(ek_f32, h_ekT, BB, NEe, DD);                   // [b,e,d] -> [b,d,e]
  gemm(h_eq, h_ekT, nullptr, nullptr, 0.f, sc_ent, nullptr,
       64, NEe, 512, 512, NEe, NEe, TT * DD, DD * NEe, TT * NEe, BB, 0);
  k_softmax_ent<<<64, 256, 0, stream>>>(sc_ent, ent_mask, h_aent);
  gemm(h_aent, h_edgeT, nullptr, nullptr, 0.f, a_para, nullptr,
       64, NSs, NEe, NEe, NSs, NSs, TT * NEe, NEe * NSs, TT * NSs, BB, 0);
  k_top5<<<2, 256, 0, stream>>>(a_para, idx_top);

  // ---- cross attention: project ALL source rows once, then gather ----
  gemm(h_srcf, wt_ca_k, ca_bk, nullptr, 0.f, nullptr, h_kall,
       (int)SR, 512, 512, 512, 512, 512, 0, 0, 0, 1, 0);
  gemm(h_srcf, wt_ca_v, ca_bv, nullptr, 0.f, nullptr, h_vall,
       (int)SR, 512, 512, 512, 512, 512, 0, 0, 0, 1, 0);
  gemm(h_qnorm, wt_ca_q, ca_bq, nullptr, 0.f, nullptr, h_cq,
       512, 512, 512, 512, 512, 512, 0, 0, 0, 1, 0);
  k_cross_attn<<<(int)BT, 320, 0, stream>>>(h_cq, h_kall, h_vall,
                                            src, idx_top, h_cctx);
  gemm(h_cctx, wt_ca_o, ca_bo, nullptr, 0.f, mid, nullptr,
       512, 512, 512, 512, 512, 512, 0, 0, 0, 1, 0);

  // ---- FFN on x = 2*mid (prescale folded into LN; residual scale 2) ----
  k_layernorm<<<(int)BT, 256, 0, stream>>>(mid, fln_g, fln_b,
                                           nullptr, h_lnx, 2.0f);
  gemm(h_lnx, wt_f1, f_b1, nullptr, 0.f, nullptr, h_h1,
       512, DFF, 512, 512, DFF, DFF, 0, 0, 0, 1, 1);            // ReLU
  gemm(h_h1, wt_f2, f_b2, mid, 2.0f, out, nullptr,
       512, 512, DFF, DFF, 512, 512, 0, 0, 0, 1, 0);            // out + 2*mid
}
